// TemporalMILBagLevel_85263690760702
// MI455X (gfx1250) — compile-verified, matrix operand order checked by reading, not run on hardware
//
#include <hip/hip_runtime.h>
#include <math.h>

#define B_    32
#define S_    64
#define H_    512
#define ATT_  128
#define NWIN  2016          // sum over widths 2..64 of (65-w) = 126*16
#define NMT   126           // M-tiles of 16 windows

typedef __attribute__((ext_vector_type(16))) __bf16 v16bf;
typedef __attribute__((ext_vector_type(8)))  float  v8f;

// ---------------------------------------------------------------------------
// Sparse-table level builder: dst[b,i,h] = max(src[b,i,h], src[b,i+half,h])
// (entries where i+half >= S equal src[b,i,h], matching -inf padding).
// ---------------------------------------------------------------------------
__global__ __launch_bounds__(256)
void build_level_kernel(const float* __restrict__ src, float* __restrict__ dst,
                        int half) {
  int idx = blockIdx.x * 256 + threadIdx.x;     // over B*S*H
  int i = (idx >> 9) & (S_ - 1);
  float v = src[idx];
  if (i + half < S_) v = fmaxf(v, src[idx + half * H_]);
  dst[idx] = v;
}

// ---------------------------------------------------------------------------
// W1 [H, ATT] fp32  ->  W1t [ATT, H] bf16  (K contiguous per output column)
// ---------------------------------------------------------------------------
__global__ __launch_bounds__(256)
void w1_transpose_bf16_kernel(const float* __restrict__ W1,
                              __bf16* __restrict__ W1t) {
  int idx = blockIdx.x * 256 + threadIdx.x;     // over H*ATT
  int h = idx / ATT_;
  int a = idx % ATT_;
  W1t[a * H_ + h] = (__bf16)W1[idx];
}

// ---------------------------------------------------------------------------
// Fused: pooled tiles -> WMMA bf16 GEMM -> relu -> W2 dot -> softmax ->
// fp32 weighted sum.  One workgroup (8 wave32) per batch.
// ---------------------------------------------------------------------------
__global__ __launch_bounds__(256)
void fused_attn_kernel(const float* __restrict__ Mst,   // [6][B][S][H] fp32
                       const __bf16* __restrict__ W1t,  // [ATT][H] bf16
                       const float* __restrict__ b1,    // [ATT]
                       const float* __restrict__ W2,    // [ATT]
                       float* __restrict__ out) {       // [B][H]
  __shared__ __bf16  sA[16][520];     // pooled tile, padded stride (bank spread)
  __shared__ float    sScore[NWIN];
  __shared__ unsigned sWin[NWIN];     // s | e2<<8 | k<<16
  __shared__ float    sRed[8];

  const int b    = blockIdx.x;
  const int tid  = threadIdx.x;
  const int lane = tid & 31;
  const int wave = tid >> 5;          // 0..7 == N-tile of ATT
  const int hf   = lane >> 4;         // half-wave
  const int r    = lane & 15;

  // Window tables + score zeroing.
  for (int n = tid; n < NWIN; n += 256) {
    int rem = n, w = 2;
    while (rem >= (S_ + 1 - w)) { rem -= (S_ + 1 - w); ++w; }
    int s  = rem;
    int k  = 31 - __clz(w);                 // floor(log2(w)), k in 1..6
    int e2 = s + w - (1 << k);
    sWin[n]   = (unsigned)s | ((unsigned)e2 << 8) | ((unsigned)k << 16);
    sScore[n] = 0.0f;
  }
  __syncthreads();

  // Hoist all W1 B-fragments for this wave's N-tile into registers.
  const int acol = wave * 16 + r;           // ATT column fed by this lane
  v16bf Bfr[16];
  for (int ks = 0; ks < 16; ++ks) {
    const __bf16* wp = W1t + (size_t)acol * H_ + ks * 32;
#pragma unroll
    for (int v = 0; v < 8; ++v) {
      int kk = (v < 4) ? ((hf ? 8 : 0) + v * 2) : ((hf ? 24 : 16) + (v - 4) * 2);
      Bfr[ks][2 * v]     = wp[kk];
      Bfr[ks][2 * v + 1] = wp[kk + 1];
    }
  }
  const float bias = b1[acol];
  const float w2v  = W2[acol];

  // ---- Pass A: scores for all 2016 windows (b2 cancels in softmax) ----
  for (int mt = 0; mt < NMT; ++mt) {
    // Build pooled tile [16 x 512] in bf16 (2 L2 reads + max per element).
    for (int idx = tid; idx < 16 * H_; idx += 256) {
      int m = idx >> 9, h = idx & (H_ - 1);
      unsigned wi = sWin[mt * 16 + m];
      int s = wi & 0xff, e2 = (wi >> 8) & 0xff, k = (int)(wi >> 16);
      const float* Mk = Mst + (((size_t)(k - 1) * B_ + b) * S_) * H_;
      sA[m][h] = (__bf16)fmaxf(Mk[s * H_ + h], Mk[e2 * H_ + h]);
    }
    __syncthreads();

    v8f c = {};
#pragma unroll
    for (int ks = 0; ks < 16; ++ks) {
      v16bf a;
      const __bf16* ap = &sA[r][ks * 32];
#pragma unroll
      for (int v = 0; v < 8; ++v) {
        int kk = (v < 4) ? ((hf ? 8 : 0) + v * 2) : ((hf ? 24 : 16) + (v - 4) * 2);
        a[2 * v]     = ap[kk];
        a[2 * v + 1] = ap[kk + 1];
      }
      c = __builtin_amdgcn_wmma_f32_16x16x32_bf16(
              false, a, false, Bfr[ks], (short)0, c, false, false);
    }

    // bias + relu + dot with W2; reduce the 16 lanes of each half-wave.
#pragma unroll
    for (int j = 0; j < 8; ++j) {
      float v = fmaxf(c[j] + bias, 0.0f) * w2v;
      v += __shfl_xor(v, 8, 16);
      v += __shfl_xor(v, 4, 16);
      v += __shfl_xor(v, 2, 16);
      v += __shfl_xor(v, 1, 16);
      if (r == 0) atomicAdd(&sScore[mt * 16 + j + hf * 8], v);
    }
    __syncthreads();
  }

  // ---- Softmax over NWIN ----
  float lm = -INFINITY;
  for (int n = tid; n < NWIN; n += 256) lm = fmaxf(lm, sScore[n]);
  for (int off = 16; off >= 1; off >>= 1) lm = fmaxf(lm, __shfl_xor(lm, off, 32));
  if (lane == 0) sRed[wave] = lm;
  __syncthreads();
  float gm = sRed[0];
  for (int i = 1; i < 8; ++i) gm = fmaxf(gm, sRed[i]);
  __syncthreads();

  float ls = 0.0f;
  for (int n = tid; n < NWIN; n += 256) {
    float e = __expf(sScore[n] - gm);
    sScore[n] = e;
    ls += e;
  }
  for (int off = 16; off >= 1; off >>= 1) ls += __shfl_xor(ls, off, 32);
  if (lane == 0) sRed[wave] = ls;
  __syncthreads();
  float gs = 0.0f;
  for (int i = 0; i < 8; ++i) gs += sRed[i];
  const float inv = 1.0f / gs;

  // ---- Pass B: fp32 weighted sum; each thread owns h = tid and tid+256 ----
  float acc0 = 0.0f, acc1 = 0.0f;
  for (int n = 0; n < NWIN; ++n) {
    unsigned wi = sWin[n];
    int s = wi & 0xff, e2 = (wi >> 8) & 0xff, k = (int)(wi >> 16);
    const float* Mk = Mst + (((size_t)(k - 1) * B_ + b) * S_) * H_;
    float p0 = fmaxf(Mk[s * H_ + tid],        Mk[e2 * H_ + tid]);
    float p1 = fmaxf(Mk[s * H_ + tid + 256],  Mk[e2 * H_ + tid + 256]);
    float w = sScore[n] * inv;
    acc0 += w * p0;
    acc1 += w * p1;
  }
  out[(size_t)b * H_ + tid]       = acc0;
  out[(size_t)b * H_ + tid + 256] = acc1;
}

// ---------------------------------------------------------------------------
extern "C" void kernel_launch(void* const* d_in, const int* in_sizes, int n_in,
                              void* d_out, int out_size, void* d_ws,
                              size_t ws_size, hipStream_t stream) {
  const float* x  = (const float*)d_in[0];   // lstm_out [B,S,H]
  const float* W1 = (const float*)d_in[1];   // [H,ATT]
  const float* b1 = (const float*)d_in[2];   // [ATT]
  const float* W2 = (const float*)d_in[3];   // [ATT,1]
  // d_in[4] (b2): constant shift before softmax -> no effect on output.

  const size_t lvl_elems = (size_t)B_ * S_ * H_;         // per level
  float*  Mst = (float*)d_ws;                            // 6 levels, 25.2 MB
  __bf16* W1t = (__bf16*)((char*)d_ws + 6 * lvl_elems * sizeof(float));

  dim3 blk(256);
  dim3 grd((unsigned)(lvl_elems / 256));

  // Level 1 from lstm_out, then levels 2..6 from previous level.
  build_level_kernel<<<grd, blk, 0, stream>>>(x, Mst, 1);
  for (int k = 2; k <= 6; ++k)
    build_level_kernel<<<grd, blk, 0, stream>>>(
        Mst + (size_t)(k - 2) * lvl_elems, Mst + (size_t)(k - 1) * lvl_elems,
        1 << (k - 1));

  w1_transpose_bf16_kernel<<<(H_ * ATT_) / 256, blk, 0, stream>>>(W1, W1t);

  fused_attn_kernel<<<B_, 256, 0, stream>>>(Mst, W1t, b1, W2, (float*)d_out);
}
